// Qwen3AttentionFused_69561290326575
// MI455X (gfx1250) — matrix operand order repaired and checked by reading, not compiled, not run on hardware
//
#include <hip/hip_runtime.h>

// ---------------------------------------------------------------------------
// Qwen3 attention block, MI455X (gfx1250, wave32, WMMA bf16, TDM).
// Pipeline: rmsnorm -> QKV GEMM (TDM X-tiles + on-the-fly f32->bf16 W staging,
//           bias folded) -> qknorm+rope(+V^T) -> flash attention (WMMA QK^T
//           and PV) -> O GEMM (+residual).
// ---------------------------------------------------------------------------

#define T_SEQ   2048
#define HIDDEN  4096
#define NH      32
#define NKV     8
#define DH      128
#define QKV_N   6144   // NH*DH + 2*NKV*DH

typedef __bf16 bf16_t;
typedef __attribute__((ext_vector_type(8)))  __bf16 v8bf;
typedef __attribute__((ext_vector_type(16))) __bf16 v16bf;
typedef __attribute__((ext_vector_type(8)))  float  v8f;
typedef __attribute__((ext_vector_type(4)))  unsigned int u32x4;
typedef __attribute__((ext_vector_type(8)))  int i32x8;
typedef __attribute__((ext_vector_type(4)))  int i32x4;

__device__ __forceinline__ v16bf cat16(v8bf lo, v8bf hi) {
  return __builtin_shufflevector(lo, hi, 0,1,2,3,4,5,6,7,8,9,10,11,12,13,14,15);
}
// A-matrix fragment (16x32 bf16, MxK): lane L -> row L&15,
// K halves sel*8+{0..7} and 16+sel*8+{0..7}   (ISA 05_wmma 7.12.2)
__device__ __forceinline__ v16bf load_a16(const bf16_t* p, int sel) {
  v8bf lo = *(const v8bf*)(p + sel*8);
  v8bf hi = *(const v8bf*)(p + 16 + sel*8);
  return cat16(lo, hi);
}
// B-matrix fragment (32x16 bf16, KxN) from row-major "W rows" storage:
// lane L -> column L&15 (= W row), K halves sel*8+{0..7} and 16+sel*8+{0..7}
__device__ __forceinline__ v16bf load_b16(const bf16_t* p, int sel) {
  v8bf lo = *(const v8bf*)(p + sel*8);
  v8bf hi = *(const v8bf*)(p + 16 + sel*8);
  return cat16(lo, hi);
}
__device__ __forceinline__ v8f wmma_bf16(v16bf a, v16bf b, v8f c) {
  return __builtin_amdgcn_wmma_f32_16x16x32_bf16(false, a, false, b,
                                                 (short)0, c, false, false);
}

// ---------------------------------------------------------------------------
// TDM: async-load one 128x32(bf16) X tile into LDS with hardware padding that
// reproduces the 40-half (80B) LDS row pitch: pad 4 DWORDs after every
// 16 DWORDs stored.  D# per CDNA5 ISA.
// X tensor: [2048, 4096] bf16 row-major (both GEMMs).
// This toolchain exposes the 6-arg builtin (amdgpu-toolchain / clang-23):
//   (u32x4 g0, i32x8 g1, i32x4 g2, i32x4 g3, i32x8 extra, i32 cpol)
// ---------------------------------------------------------------------------
__device__ __forceinline__ void tdm_load_x_tile(const bf16_t* X, long mBase,
                                                int kcol, const bf16_t* ldsdst) {
  const unsigned long long ga =
      (unsigned long long)(uintptr_t)(X + (unsigned long long)mBase * HIDDEN + kcol);
  const unsigned la = (unsigned)(uintptr_t)ldsdst;
  // group0: count=1 | lds_addr | global_addr[56:0] | type=2 ("image")
  u32x4 g0 = { 1u, la, (unsigned)ga,
               (unsigned)((ga >> 32) & 0x1FFFFFFull) | (2u << 30) };
  // group1: data_size=2B(1<<16) | pad_enable(1<<20) | pad_interval=16dw(3<<22)
  //         | pad_amount=4dw(3<<25); tensor_dim0=4096; tensor_dim1=2048;
  //         tile_dim0=32; tile_dim1=128; tensor_dim0_stride=4096
  i32x8 g1 = { (int)0x06D10000,
               (int)(((unsigned)HIDDEN & 0xFFFFu) << 16),   // dim0 low16 @ [31:16]
               (int)(((unsigned)T_SEQ  & 0xFFFFu) << 16),   // dim1 low16 @ [31:16]
               (int)(32u << 16),                            // tile_dim0 @ [31:16]
               128,                                         // tile_dim1
               HIDDEN,                                      // dim0 stride
               0, 0 };
  i32x4 gz4 = { 0, 0, 0, 0 };
  i32x8 gz8 = { 0, 0, 0, 0, 0, 0, 0, 0 };
  __builtin_amdgcn_tensor_load_to_lds(g0, g1, gz4, gz4, gz8, 0);
}

// ---------------------------------------------------------------------------
// RMSNorm(hidden) * ln_weight -> bf16
// ---------------------------------------------------------------------------
__global__ __launch_bounds__(256) void rmsnorm_kernel(
    const float* __restrict__ H, const float* __restrict__ Wt,
    bf16_t* __restrict__ Out) {
  const int t = blockIdx.x, tid = threadIdx.x;
  const int lane = tid & 31, wid = tid >> 5;
  __shared__ float red[8];
  const float* row = H + (long)t * HIDDEN;
  float ss = 0.f;
  for (int i = tid; i < HIDDEN; i += 256) { const float x = row[i]; ss += x * x; }
  #pragma unroll
  for (int mm = 16; mm >= 1; mm >>= 1) ss += __shfl_xor(ss, mm, 32);
  if (lane == 0) red[wid] = ss;
  __syncthreads();
  float tot = 0.f;
  #pragma unroll
  for (int w = 0; w < 8; ++w) tot += red[w];
  const float rs = rsqrtf(tot * (1.f / (float)HIDDEN) + 1e-6f);
  for (int i = tid; i < HIDDEN; i += 256)
    Out[(long)t * HIDDEN + i] = (bf16_t)(row[i] * rs * Wt[i]);
}

// ---------------------------------------------------------------------------
// GEMM: C[M,NTOT] = X[M,4096](bf16) @ Wf[NTOT,4096](f32, converted on the fly)^T
//       (+bias init | +residual epilogue)
// Block 128x128, 8 waves as 2(M)x4(N), wave tile 64x32, K-step 32.
// Double-buffered LDS; X tiles via TDM (TENSORcnt), W tiles via VALU staging
// (f32 loads issued before compute, converted+stored after -> latency hidden).
// ---------------------------------------------------------------------------
template <int NTOT, bool ADD_RESID>
__global__ __launch_bounds__(256) void gemm_kernel(
    const bf16_t* __restrict__ X, const float* __restrict__ Wf,
    const float* __restrict__ aux, float* __restrict__ C) {
  constexpr int LDW = 40;           // halves per LDS row (80B, conflict-free)
  __shared__ __align__(32) bf16_t Xs[2][128 * LDW];
  __shared__ __align__(32) bf16_t Wsh[2][128 * LDW];
  const int tid = threadIdx.x, lane = tid & 31, wid = tid >> 5;
  const int wm = wid >> 2, wn = wid & 3;
  const int sel = lane >> 4, ln = lane & 15;
  const long mBase = (long)blockIdx.y * 128;
  const long nBase = (long)blockIdx.x * 128;

  v8f acc[4][2];
  if (!ADD_RESID) {  // fold qkv_bias into accumulator init (bias is per-N)
    #pragma unroll
    for (int j = 0; j < 2; ++j) {
      const float b = aux[nBase + wn * 32 + j * 16 + ln];
      #pragma unroll
      for (int i = 0; i < 4; ++i)
        #pragma unroll
        for (int e = 0; e < 8; ++e) acc[i][j][e] = b;
    }
  } else {
    #pragma unroll
    for (int i = 0; i < 4; ++i)
      #pragma unroll
      for (int j = 0; j < 2; ++j)
        #pragma unroll
        for (int e = 0; e < 8; ++e) acc[i][j][e] = 0.f;
  }

  const int lrow = tid >> 1, lhalf = (tid & 1) * 16;  // 16 elems per thread
  const float* gWf = Wf + (long)(nBase + lrow) * HIDDEN + lhalf;
  bf16_t* sWbase[2] = { Wsh[0] + lrow * LDW + lhalf, Wsh[1] + lrow * LDW + lhalf };

  constexpr int NK = HIDDEN / 32;

  // ---- prologue: tile 0 ----
  if (wid == 0) tdm_load_x_tile(X, mBase, 0, Xs[0]);
  {
    float wreg[16];
    #pragma unroll
    for (int e = 0; e < 16; ++e) wreg[e] = gWf[e];
    alignas(16) bf16_t tmp[16];
    #pragma unroll
    for (int e = 0; e < 16; ++e) tmp[e] = (bf16_t)wreg[e];
    *(v8bf*)(sWbase[0])     = *(v8bf*)&tmp[0];
    *(v8bf*)(sWbase[0] + 8) = *(v8bf*)&tmp[8];
  }
  if (wid == 0) __builtin_amdgcn_s_wait_tensorcnt(0);
  __syncthreads();

  for (int kt = 0; kt < NK; ++kt) {
    const int cur = kt & 1, nxt = cur ^ 1;
    const bool more = (kt + 1) < NK;

    // issue next X tile DMA + next W f32 loads before computing current tile
    if (more && wid == 0) tdm_load_x_tile(X, mBase, (kt + 1) * 32, Xs[nxt]);
    float wreg[16];
    if (more) {
      const float* src = gWf + (kt + 1) * 32;
      #pragma unroll
      for (int e = 0; e < 16; ++e) wreg[e] = src[e];
    }

    // compute tile kt
    v16bf a[4], b[2];
    #pragma unroll
    for (int i = 0; i < 4; ++i)
      a[i] = load_a16(Xs[cur] + (wm * 64 + i * 16 + ln) * LDW, sel);
    #pragma unroll
    for (int j = 0; j < 2; ++j)
      b[j] = load_b16(Wsh[cur] + (wn * 32 + j * 16 + ln) * LDW, sel);
    #pragma unroll
    for (int i = 0; i < 4; ++i)
      #pragma unroll
      for (int j = 0; j < 2; ++j)
        acc[i][j] = wmma_bf16(a[i], b[j], acc[i][j]);

    // convert + store next W tile into alternate buffer
    if (more) {
      alignas(16) bf16_t tmp[16];
      #pragma unroll
      for (int e = 0; e < 16; ++e) tmp[e] = (bf16_t)wreg[e];
      *(v8bf*)(sWbase[nxt])     = *(v8bf*)&tmp[0];
      *(v8bf*)(sWbase[nxt] + 8) = *(v8bf*)&tmp[8];
    }
    if (more && wid == 0) __builtin_amdgcn_s_wait_tensorcnt(0);
    __syncthreads();
  }

  // C/D layout: VGPR r, lane L -> M = r + 8*(L>>4), N = L&15
  #pragma unroll
  for (int j = 0; j < 2; ++j) {
    const long n = nBase + wn * 32 + j * 16 + ln;
    #pragma unroll
    for (int i = 0; i < 4; ++i) {
      const long m0 = mBase + wm * 64 + i * 16 + 8 * sel;
      #pragma unroll
      for (int r = 0; r < 8; ++r) {
        const long idx = (m0 + r) * NTOT + n;
        float v = acc[i][j][r];
        if (ADD_RESID) v += aux[idx];
        C[idx] = v;
      }
    }
  }
}

// ---------------------------------------------------------------------------
// per-head RMSNorm(q,k) + RoPE -> bf16; V -> bf16 transposed [NKV, D, T]
// ---------------------------------------------------------------------------
__global__ __launch_bounds__(128) void qknorm_rope_kernel(
    const float* __restrict__ QKV, const int* __restrict__ positions,
    const float* __restrict__ qw, const float* __restrict__ kw,
    bf16_t* __restrict__ Qo, bf16_t* __restrict__ Ko, bf16_t* __restrict__ VTo) {
  const int t = blockIdx.x, tid = threadIdx.x;
  const int lane = tid & 31, wid = tid >> 5;
  __shared__ float red[4];
  __shared__ float xs[DH];
  const float pos = (float)positions[t];
  const float* row = QKV + (long)t * QKV_N;
  const int fi = tid & 63;                       // rope frequency index
  const float inv_freq = exp2f(-(float)fi * (13.2877123795f / 64.f)); // 10000^(-fi/64)
  float sn, cs;
  __sincosf(pos * inv_freq, &sn, &cs);

  for (int hh = 0; hh < NH + NKV; ++hh) {
    const bool isq = hh < NH;
    const int off = isq ? hh * DH : HIDDEN + (hh - NH) * DH;
    const float x = row[off + tid];
    float ss = x * x;
    #pragma unroll
    for (int mm = 16; mm >= 1; mm >>= 1) ss += __shfl_xor(ss, mm, 32);
    if (lane == 0) red[wid] = ss;
    __syncthreads();
    const float tot = red[0] + red[1] + red[2] + red[3];
    const float rs = rsqrtf(tot * (1.f / (float)DH) + 1e-6f);
    const float xn = x * rs * (isq ? qw[tid] : kw[tid]);
    xs[tid] = xn;
    __syncthreads();
    const float other = xs[tid ^ 64];
    const float outv = (tid < 64) ? (xn * cs - other * sn)
                                  : (xn * cs + other * sn);
    if (isq) Qo[((long)t * NH  + hh)        * DH + tid] = (bf16_t)outv;
    else     Ko[((long)t * NKV + (hh - NH)) * DH + tid] = (bf16_t)outv;
    __syncthreads();
  }
  #pragma unroll
  for (int kv = 0; kv < NKV; ++kv) {
    const float v = row[HIDDEN + NKV * DH + kv * DH + tid];
    VTo[((long)(kv * DH + tid)) * T_SEQ + t] = (bf16_t)v;   // V^T for PV B-frags
  }
}

// ---------------------------------------------------------------------------
// Flash attention: one head per blockIdx.y, 4 waves x 16 query rows.
// QK^T and PV both via v_wmma_f32_16x16x32_bf16; K/V frags from L2.
// ---------------------------------------------------------------------------
__global__ __launch_bounds__(128) void attn_kernel(
    const bf16_t* __restrict__ Q, const bf16_t* __restrict__ K,
    const bf16_t* __restrict__ VT, bf16_t* __restrict__ O) {
  const int h = blockIdx.y;
  const int kh = h >> 2;                 // GQA: 4 q-heads per kv-head
  const int tid = threadIdx.x, wid = tid >> 5, lane = tid & 31;
  const int sel = lane >> 4, ln = lane & 15;
  const int q0 = blockIdx.x * 64 + wid * 16;

  __shared__ __align__(32) bf16_t p_lds[4][16 * 32];   // wave-private P staging
  bf16_t* P = p_lds[wid];

  // Q fragments (16 rows x 128 d), register-resident for the whole row tile
  v16bf qf[4];
  {
    const bf16_t* qrow = Q + ((long)(q0 + ln) * NH + h) * DH;
    #pragma unroll
    for (int s2 = 0; s2 < 4; ++s2) qf[s2] = load_a16(qrow + s2 * 32, sel);
  }

  v8f o_acc[8];
  #pragma unroll
  for (int d2 = 0; d2 < 8; ++d2)
    #pragma unroll
    for (int e = 0; e < 8; ++e) o_acc[d2][e] = 0.f;
  float mrow[8], lrow[8];
  #pragma unroll
  for (int r = 0; r < 8; ++r) { mrow[r] = -1e30f; lrow[r] = 0.f; }

  const float scale = 0.08838834764831845f;     // D^-0.5
  const int kend = q0 + 16;                      // causal: keys <= q0+15

  for (int kb0 = 0; kb0 < kend; kb0 += 32) {
    v8f s0, s1;
    #pragma unroll
    for (int e = 0; e < 8; ++e) { s0[e] = 0.f; s1[e] = 0.f; }
    const bf16_t* kr0 = K + ((long)(kb0 + ln)      * NKV + kh) * DH;
    const bf16_t* kr1 = K + ((long)(kb0 + 16 + ln) * NKV + kh) * DH;
    #pragma unroll
    for (int s2 = 0; s2 < 4; ++s2) {
      v16bf b0 = load_b16(kr0 + s2 * 32, sel);
      v16bf b1 = load_b16(kr1 + s2 * 32, sel);
      s0 = wmma_bf16(qf[s2], b0, s0);
      s1 = wmma_bf16(qf[s2], b1, s1);
    }
    // online softmax over the 32-key tile (row groups = 16-lane halves)
    float p0[8], p1[8], frow[8];
    const int c0 = kb0 + ln, c1 = c0 + 16;
    #pragma unroll
    for (int r = 0; r < 8; ++r) {
      const int tq = q0 + r + 8 * sel;
      float v0 = (c0 <= tq) ? s0[r] * scale : -1e30f;
      float v1 = (c1 <= tq) ? s1[r] * scale : -1e30f;
      float mx = fmaxf(v0, v1);
      #pragma unroll
      for (int mm = 8; mm >= 1; mm >>= 1) mx = fmaxf(mx, __shfl_xor(mx, mm, 32));
      const float mn = fmaxf(mrow[r], mx);
      frow[r] = __expf(mrow[r] - mn);
      mrow[r] = mn;
      p0[r] = __expf(v0 - mn);
      p1[r] = __expf(v1 - mn);
      float sum = p0[r] + p1[r];
      #pragma unroll
      for (int mm = 8; mm >= 1; mm >>= 1) sum += __shfl_xor(sum, mm, 32);
      lrow[r] = lrow[r] * frow[r] + sum;
    }
    #pragma unroll
    for (int d2 = 0; d2 < 8; ++d2)
      #pragma unroll
      for (int r = 0; r < 8; ++r) o_acc[d2][r] *= frow[r];
    // C-layout -> A-layout via LDS (same-wave DS ops are in-order; no barrier)
    #pragma unroll
    for (int r = 0; r < 8; ++r) {
      const int prow = r + 8 * sel;
      P[prow * 32 + ln]      = (bf16_t)p0[r];
      P[prow * 32 + 16 + ln] = (bf16_t)p1[r];
    }
    v16bf pa = load_a16(P + ln * 32, sel);
    #pragma unroll
    for (int d2 = 0; d2 < 8; ++d2) {
      const bf16_t* vr = VT + ((long)(kh * DH + d2 * 16 + ln)) * T_SEQ + kb0;
      v16bf bv = load_b16(vr, sel);
      o_acc[d2] = wmma_bf16(pa, bv, o_acc[d2]);
    }
  }

  #pragma unroll
  for (int r = 0; r < 8; ++r) lrow[r] = 1.f / lrow[r];
  #pragma unroll
  for (int d2 = 0; d2 < 8; ++d2)
    #pragma unroll
    for (int r = 0; r < 8; ++r) {
      const int tq = q0 + r + 8 * sel;
      O[(long)tq * HIDDEN + h * DH + d2 * 16 + ln] =
          (bf16_t)(o_acc[d2][r] * lrow[r]);
    }
}

// ---------------------------------------------------------------------------
// launch
// ---------------------------------------------------------------------------
extern "C" void kernel_launch(void* const* d_in, const int* in_sizes, int n_in,
                              void* d_out, int out_size, void* d_ws, size_t ws_size,
                              hipStream_t stream) {
  (void)in_sizes; (void)n_in; (void)out_size; (void)ws_size;
  const int*   positions = (const int*)d_in[0];
  const float* hidden    = (const float*)d_in[1];
  const float* lnw       = (const float*)d_in[2];
  const float* resid     = (const float*)d_in[3];
  const float* qkvw      = (const float*)d_in[4];
  const float* qkvb      = (const float*)d_in[5];
  const float* qnw       = (const float*)d_in[6];
  const float* knw       = (const float*)d_in[7];
  const float* ow        = (const float*)d_in[8];
  float* out = (float*)d_out;

  char* ws = (char*)d_ws;
  size_t off = 0;
  bf16_t* normed = (bf16_t*)(ws + off); off += (size_t)T_SEQ * HIDDEN * 2;   // 16 MiB
  float*  qkvf   = (float*) (ws + off); off += (size_t)T_SEQ * QKV_N * 4;    // 48 MiB
  bf16_t* qbuf   = (bf16_t*)(ws + off); off += (size_t)T_SEQ * NH * DH * 2;  // 16 MiB
  bf16_t* kbuf   = (bf16_t*)(ws + off); off += (size_t)T_SEQ * NKV * DH * 2; //  4 MiB
  bf16_t* vtb    = (bf16_t*)(ws + off); off += (size_t)NKV * DH * T_SEQ * 2; //  4 MiB
  bf16_t* attnb  = (bf16_t*)(ws + off); off += (size_t)T_SEQ * HIDDEN * 2;   // 16 MiB

  rmsnorm_kernel<<<T_SEQ, 256, 0, stream>>>(hidden, lnw, normed);
  gemm_kernel<QKV_N, false><<<dim3(QKV_N / 128, T_SEQ / 128), 256, 0, stream>>>(
      normed, qkvw, qkvb, qkvf);
  qknorm_rope_kernel<<<T_SEQ, 128, 0, stream>>>(qkvf, positions, qnw, knw,
                                                qbuf, kbuf, vtb);
  attn_kernel<<<dim3(T_SEQ / 64, NH), 128, 0, stream>>>(qbuf, kbuf, vtb, attnb);
  gemm_kernel<HIDDEN, true><<<dim3(HIDDEN / 128, T_SEQ / 128), 256, 0, stream>>>(
      attnb, ow, resid, out);
}
